// MotionVQVAE_45741401702607
// MI455X (gfx1250) — matrix operand order repaired and checked by reading, not compiled, not run on hardware
//
#include <hip/hip_runtime.h>

// ---------------------------------------------------------------------------
// Types for CDNA5 WMMA (wave32): bf16 A/B fragments, f32 accumulator.
// ---------------------------------------------------------------------------
typedef __bf16 bf16_t;
typedef bf16_t v16bf __attribute__((ext_vector_type(16)));
typedef bf16_t v8bf  __attribute__((ext_vector_type(8)));
typedef float  v8f   __attribute__((ext_vector_type(8)));

static __device__ __forceinline__ unsigned short f32_to_bf16_rne(float f) {
  union { float f; unsigned u; } v; v.f = f;
  unsigned r = v.u + 0x7fffu + ((v.u >> 16) & 1u);
  return (unsigned short)(r >> 16);
}

// ---------------------------------------------------------------------------
// Weight packing: fp32 conv weights -> bf16 rows [O][KKpad], KKpad = ceil32(I*K).
// For ConvTranspose1d (torch [I,O,K]) fold the flip+transpose: w_eff[o][i][k] =
// w[i][o][K-1-k], so the conv kernel below runs it as a plain dilated conv.
// ---------------------------------------------------------------------------
__global__ __launch_bounds__(256) void prep_weight_kernel(
    const float* __restrict__ w, unsigned short* __restrict__ out,
    int O, int I, int Kk, int KKpad, int isT) {
  const int idx = blockIdx.x * 256 + threadIdx.x;
  if (idx >= O * KKpad) return;
  const int o = idx / KKpad;
  const int r = idx - o * KKpad;
  unsigned short v = 0;
  if (r < I * Kk) {
    const int ci = r / Kk;
    const int k  = r - ci * Kk;
    const float f = isT ? w[((size_t)ci * O + o) * Kk + (Kk - 1 - k)]
                        : w[((size_t)o * I + ci) * Kk + k];
    v = f32_to_bf16_rne(f);
  }
  out[idx] = v;
}

__global__ __launch_bounds__(256) void prep_codebook_kernel(
    const float* __restrict__ cb, unsigned short* __restrict__ cbq,
    float* __restrict__ cnorm) {
  const int k = blockIdx.x * 256 + threadIdx.x;
  if (k >= 512) return;
  float s = 0.0f;
  for (int d = 0; d < 512; ++d) {
    const float c = cb[(size_t)k * 512 + d];
    cbq[(size_t)k * 512 + d] = f32_to_bf16_rne(c);
    s += c * c;
  }
  cnorm[k] = s;
}

// ---------------------------------------------------------------------------
// Generic implicit-GEMM conv1d (Conv1d and ConvTranspose1d-as-dilated-conv).
// Block = 256 threads = 8 waves; block tile = 256 out-channels x 32 time of
// one batch. Each wave owns 32 channels x 32 time: 2 A fragments x 2 B
// fragments -> 4 WMMAs per reduction chunk, full operand cross-reuse. The
// 32x32 bf16 im2col patch is staged once per chunk in LDS, shared by all 8
// waves. A per-block LDS table packs ((ci*Ti)<<4 | invalid<<3 | tap) so the
// hot fill loop is pure shift/add + one predicated global load — no integer
// division, no exec-mask branching (addresses clamp to 0, values select 0).
// stride / lhs-dilation are passed as shifts (values are only 1 or 2).
// ---------------------------------------------------------------------------
struct ConvArgs {
  const float* srcf; const unsigned short* src16;
  const float* bias; const unsigned short* wq;
  float* dstf; unsigned short* dst16;
  int I, Ti, O, To, Kk, KKpad, padE;
  int sh;           // log2(stride)       (0 or 1)
  int dl;           // log2(lhs_dilation) (0 or 1)
  int src_f32btf;   // 1: src is fp32 [B,T,F]; 0: src is bf16 [B,C,T]
  int dst_fmt;      // 0: bf16 [B,C,T]; 1: fp32 [B,C,T]; 2: fp32 [B,T,F]
  int relu;
};

__global__ __launch_bounds__(256) void conv1d_wmma_kernel(ConvArgs a) {
  const int t0    = blockIdx.x << 5;            // 32-wide time tile
  const int b     = blockIdx.y;
  const int tid   = threadIdx.x;
  const int lane  = tid & 31;
  const int laneN = lane & 15;
  const int hi    = (lane >> 4) & 1;
  const int oBase = (blockIdx.z << 8) + ((tid >> 5) << 5);  // 32 channels/wave

  __shared__ __align__(16) unsigned short patch[32][32];  // [time n][reduction r]
  __shared__ int rtab[2048];  // ((ci*Ti)<<4)|inv<<3|tap  (or ci<<4|... for BTF)

  // One-time reduction-index decode table (no div/mod in the fill loop).
  {
    const int IK = a.I * a.Kk;
    for (int r = tid; r < a.KKpad; r += 256) {
      const int ci  = r / a.Kk;
      const int k   = r - ci * a.Kk;
      const int inv = (r >= IK) ? 8 : 0;
      const int q   = a.src_f32btf ? ci : ci * a.Ti;
      rtab[r] = (q << 4) | inv | k;
    }
  }

  v8f acc00, acc01, acc10, acc11;
#pragma unroll
  for (int i = 0; i < 8; ++i) {
    acc00[i] = 0.0f; acc01[i] = 0.0f; acc10[i] = 0.0f; acc11[i] = 0.0f;
  }

  const int nChunks   = a.KKpad >> 5;
  const unsigned lim  = (unsigned)(a.Ti << a.dl);
  const int dmask     = (1 << a.dl) - 1;
  const size_t bbase  = (size_t)b * a.I * a.Ti;
  const unsigned short* wrow0 =
      a.wq + (size_t)(oBase + laneN) * a.KKpad + (hi << 3);
  const unsigned short* wrow1 = wrow0 + (size_t)16 * a.KKpad;
  __syncthreads();  // rtab ready

  for (int kc = 0; kc < nChunks; ++kc) {
    // ---- cooperatively stage im2col tile (32 time x 32 reduction) ----
    for (int e = tid; e < 1024; e += 256) {
      const int n  = e >> 5;
      const int r  = e & 31;
      const int tq = rtab[(kc << 5) + r];
      const int k  = tq & 7;
      const int p  = ((t0 + n) << a.sh) + k - a.padE;
      const bool ok = ((unsigned)p < lim) & ((p & dmask) == 0) & ((tq & 8) == 0);
      unsigned short v;
      if (a.src_f32btf) {
        const int ci = tq >> 4;
        const size_t idx = ok ? ((size_t)(b * a.Ti + p) * a.I + ci) : 0;
        const float f = a.srcf[idx];
        v = ok ? f32_to_bf16_rne(f) : (unsigned short)0;
      } else {
        const size_t idx = ok ? (bbase + (size_t)(tq >> 4) + (p >> a.dl)) : 0;
        const unsigned short lv = a.src16[idx];
        v = ok ? lv : (unsigned short)0;
      }
      patch[n][r] = v;
    }
    __syncthreads();

    // ---- A fragments: packed bf16 weights (16-bit A 16x32 lane layout) ----
    const unsigned short* wr0 = wrow0 + (kc << 5);
    const unsigned short* wr1 = wrow1 + (kc << 5);
    __builtin_prefetch(wr0 + 32, 0, 1);
    __builtin_prefetch(wr1 + 32, 0, 1);
    v8bf a00 = *(const v8bf*)(wr0);
    v8bf a01 = *(const v8bf*)(wr0 + 16);
    v8bf a10 = *(const v8bf*)(wr1);
    v8bf a11 = *(const v8bf*)(wr1 + 16);
    v16bf af0, af1;
#pragma unroll
    for (int i = 0; i < 8; ++i) {
      af0[i] = a00[i]; af0[i + 8] = a01[i];
      af1[i] = a10[i]; af1[i + 8] = a11[i];
    }

    // ---- B fragments from LDS (two 16-wide time halves) ----
    const unsigned short* zr0 = &patch[laneN][hi << 3];
    const unsigned short* zr1 = &patch[laneN + 16][hi << 3];
    v8bf b00 = *(const v8bf*)(zr0);
    v8bf b01 = *(const v8bf*)(zr0 + 16);
    v8bf b10 = *(const v8bf*)(zr1);
    v8bf b11 = *(const v8bf*)(zr1 + 16);
    v16bf bf0, bf1;
#pragma unroll
    for (int i = 0; i < 8; ++i) {
      bf0[i] = b00[i]; bf0[i + 8] = b01[i];
      bf1[i] = b10[i]; bf1[i + 8] = b11[i];
    }

    acc00 = __builtin_amdgcn_wmma_f32_16x16x32_bf16(false, af0, false, bf0,
                                                    (short)0, acc00, false, false);
    acc01 = __builtin_amdgcn_wmma_f32_16x16x32_bf16(false, af0, false, bf1,
                                                    (short)0, acc01, false, false);
    acc10 = __builtin_amdgcn_wmma_f32_16x16x32_bf16(false, af1, false, bf0,
                                                    (short)0, acc10, false, false);
    acc11 = __builtin_amdgcn_wmma_f32_16x16x32_bf16(false, af1, false, bf1,
                                                    (short)0, acc11, false, false);
    __syncthreads();
  }

  // ---- epilogue: C/D layout (VGPR j: M=j(+8 for hi lanes), N=laneN) ----
  const int mh = hi << 3;
  const int tA = t0 + laneN;
  const int tB = tA + 16;
#pragma unroll
  for (int half = 0; half < 2; ++half) {
    const v8f& aA = half ? acc10 : acc00;
    const v8f& aB = half ? acc11 : acc01;
#pragma unroll
    for (int j = 0; j < 8; ++j) {
      const int o = oBase + (half << 4) + j + mh;
      const float bv = a.bias[o];
      float v0 = aA[j] + bv;
      float v1 = aB[j] + bv;
      if (a.relu) { v0 = fmaxf(v0, 0.0f); v1 = fmaxf(v1, 0.0f); }
      if (a.dst_fmt == 0) {
        const size_t base = ((size_t)b * a.O + o) * a.To;
        a.dst16[base + tA] = f32_to_bf16_rne(v0);
        a.dst16[base + tB] = f32_to_bf16_rne(v1);
      } else if (a.dst_fmt == 1) {
        const size_t base = ((size_t)b * a.O + o) * a.To;
        a.dstf[base + tA] = v0;
        a.dstf[base + tB] = v1;
      } else {
        a.dstf[((size_t)b * a.To + tA) * a.O + o] = v0;
        a.dstf[((size_t)b * a.To + tB) * a.O + o] = v1;
      }
    }
  }
}

// ---------------------------------------------------------------------------
// VQ argmin: per block one 16-position z-tile (bf16 in LDS, 16KB), 8 waves
// sweep all 512 codes, WMMA dot products, score = |c|^2 - 2 z.c, deterministic
// first-min argmin reduction through LDS.
// ---------------------------------------------------------------------------
__global__ __launch_bounds__(256) void vq_argmin_kernel(
    const float* __restrict__ zbuf, const unsigned short* __restrict__ cbq,
    const float* __restrict__ cnorm, float* __restrict__ idx_f,
    int* __restrict__ idx_i, int Tq) {
  const int t0    = blockIdx.x << 4;
  const int b     = blockIdx.y;
  const int tid   = threadIdx.x;
  const int lane  = tid & 31;
  const int laneN = lane & 15;
  const int hi    = (lane >> 4) & 1;
  const int wv    = tid >> 5;

  __shared__ __align__(16) unsigned short zlds[16][512];
  __shared__ float pv[256];
  __shared__ int   pi[256];

  for (int e = tid; e < 16 * 512; e += 256) {
    const int n = e >> 9;
    const int d = e & 511;
    zlds[n][d] = f32_to_bf16_rne(zbuf[((size_t)b * 512 + d) * Tq + (t0 + n)]);
  }
  __syncthreads();

  float minv = 3.402823466e38f;
  int   mink = 0;

  for (int ct = 0; ct < 4; ++ct) {          // 512 codes / (8 waves * 16)
    const int kBase = (ct << 7) + (wv << 4);
    v8f acc;
#pragma unroll
    for (int i = 0; i < 8; ++i) acc[i] = 0.0f;

    for (int kc = 0; kc < 16; ++kc) {       // D=512 / 32
      const unsigned short* cr =
          cbq + (size_t)(kBase + laneN) * 512 + (kc << 5) + (hi << 3);
      __builtin_prefetch(cr + 32, 0, 1);
      v8bf a0 = *(const v8bf*)(cr);
      v8bf a1 = *(const v8bf*)(cr + 16);
      v16bf afrag;
#pragma unroll
      for (int i = 0; i < 8; ++i) { afrag[i] = a0[i]; afrag[i + 8] = a1[i]; }

      const unsigned short* zr = &zlds[laneN][(kc << 5) + (hi << 3)];
      v8bf b0 = *(const v8bf*)(zr);
      v8bf b1 = *(const v8bf*)(zr + 16);
      v16bf bfrag;
#pragma unroll
      for (int i = 0; i < 8; ++i) { bfrag[i] = b0[i]; bfrag[i + 8] = b1[i]; }

      acc = __builtin_amdgcn_wmma_f32_16x16x32_bf16(false, afrag, false, bfrag,
                                                    (short)0, acc, false, false);
    }
#pragma unroll
    for (int j = 0; j < 8; ++j) {
      const int k = kBase + j + (hi << 3);
      const float s = cnorm[k] - 2.0f * acc[j];
      if (s < minv || (s == minv && k < mink)) { minv = s; mink = k; }
    }
  }

  pv[tid] = minv; pi[tid] = mink;
  __syncthreads();

  if (tid < 16) {                           // one reducer per time position
    float bv = 3.402823466e38f; int bi = 0;
    for (int w = 0; w < 8; ++w)
      for (int h = 0; h < 32; h += 16) {
        const int e = (w << 5) + tid + h;   // lanes with laneN == tid
        const float v = pv[e]; const int ix = pi[e];
        if (v < bv || (v == bv && ix < bi)) { bv = v; bi = ix; }
      }
    const int gi = b * Tq + t0 + tid;
    idx_f[gi] = (float)bi;
    idx_i[gi] = bi;
  }
}

// ---------------------------------------------------------------------------
// Gather z_q = codebook[idx] (bf16, decoder input) and per-position squared
// error partials in fp32. Fixed-order reductions -> bitwise deterministic.
// ---------------------------------------------------------------------------
__global__ __launch_bounds__(256) void vq_gather_loss_kernel(
    const float* __restrict__ zbuf, const float* __restrict__ cb,
    const int* __restrict__ idx_i, unsigned short* __restrict__ zq16,
    float* __restrict__ partial, int Tq) {
  const int n   = blockIdx.x;               // b*Tq + t
  const int b   = n / Tq;
  const int t   = n - b * Tq;
  const int tid = threadIdx.x;
  const int code = idx_i[n];
  __shared__ float red[256];
  float s = 0.0f;
  for (int d = tid; d < 512; d += 256) {
    const float c  = cb[(size_t)code * 512 + d];
    const size_t zi = ((size_t)b * 512 + d) * Tq + t;
    const float z  = zbuf[zi];
    zq16[zi] = f32_to_bf16_rne(c);
    const float df = z - c;
    s += df * df;
  }
  red[tid] = s; __syncthreads();
  for (int off = 128; off > 0; off >>= 1) {
    if (tid < off) red[tid] += red[tid + off];
    __syncthreads();
  }
  if (tid == 0) partial[n] = red[0];
}

__global__ __launch_bounds__(256) void vq_loss_finalize_kernel(
    const float* __restrict__ partial, float* __restrict__ out, int N) {
  const int tid = threadIdx.x;
  __shared__ float red[256];
  float s = 0.0f;
  for (int i = tid; i < N; i += 256) s += partial[i];
  red[tid] = s; __syncthreads();
  for (int off = 128; off > 0; off >>= 1) {
    if (tid < off) red[tid] += red[tid + off];
    __syncthreads();
  }
  if (tid == 0) out[0] = 0.25f * red[0] / ((float)N * 512.0f);
}

// ---------------------------------------------------------------------------
// Host orchestration.
// ---------------------------------------------------------------------------
extern "C" void kernel_launch(void* const* d_in, const int* in_sizes, int n_in,
                              void* d_out, int out_size, void* d_ws, size_t ws_size,
                              hipStream_t stream) {
  (void)in_sizes; (void)n_in; (void)out_size; (void)ws_size;
  const int B = 64, T = 512, F = 256, D = 512, Tq = 128, NC = 512;

  const float* x    = (const float*)d_in[0];
  const float* ew1  = (const float*)d_in[1];  const float* eb1  = (const float*)d_in[2];
  const float* ew2  = (const float*)d_in[3];  const float* eb2  = (const float*)d_in[4];
  const float* ew3  = (const float*)d_in[5];  const float* eb3  = (const float*)d_in[6];
  const float* ew4  = (const float*)d_in[7];  const float* eb4  = (const float*)d_in[8];
  const float* ew5  = (const float*)d_in[9];  const float* eb5  = (const float*)d_in[10];
  const float* cb   = (const float*)d_in[11];
  const float* dw1  = (const float*)d_in[12]; const float* db1  = (const float*)d_in[13];
  const float* dwt2 = (const float*)d_in[14]; const float* dbt2 = (const float*)d_in[15];
  const float* dw3  = (const float*)d_in[16]; const float* db3  = (const float*)d_in[17];
  const float* dwt4 = (const float*)d_in[18]; const float* dbt4 = (const float*)d_in[19];
  const float* dw5  = (const float*)d_in[20]; const float* db5  = (const float*)d_in[21];

  float* outp  = (float*)d_out;
  float* recon = outp;                         // B*T*F
  float* idx_f = outp + (size_t)B * T * F;     // B*Tq (indices as float)
  float* loss  = idx_f + (size_t)B * Tq;       // 1

  // Workspace carving (256B aligned).
  char* wsb = (char*)d_ws;
  size_t cur = 0;
  auto carve = [&](size_t bytes) -> void* {
    void* p = wsb + cur;
    cur = (cur + bytes + 255) & ~(size_t)255;
    return p;
  };
  unsigned short* bufA = (unsigned short*)carve((size_t)8388608 * 2);  // bf16 acts
  unsigned short* bufB = (unsigned short*)carve((size_t)8388608 * 2);
  float* zbuf  = (float*)carve((size_t)B * D * Tq * 4);                // fp32 z
  unsigned short* zq16 = (unsigned short*)carve((size_t)B * D * Tq * 2);
  int*   idxi  = (int*)carve((size_t)B * Tq * 4);
  float* part  = (float*)carve((size_t)B * Tq * 4);
  float* cnorm = (float*)carve((size_t)NC * 4);
  unsigned short* cbq = (unsigned short*)carve((size_t)NC * D * 2);

  // Pack all conv weights to bf16 A-matrix layout.
  struct WSpec { const float* w; int O, I, Kk, isT; };
  const WSpec wspec[10] = {
    {ew1, 256, 256, 3, 0}, {ew2, 256, 256, 4, 0}, {ew3, 512, 256, 3, 0},
    {ew4, 512, 512, 4, 0}, {ew5, 512, 512, 3, 0},
    {dw1, 512, 512, 3, 0}, {dwt2, 512, 512, 4, 1},
    {dw3, 256, 512, 3, 0}, {dwt4, 256, 256, 4, 1}, {dw5, 256, 256, 3, 0},
  };
  unsigned short* wq[10];
  int kkpad[10];
  for (int i = 0; i < 10; ++i) {
    const int KK = wspec[i].I * wspec[i].Kk;
    kkpad[i] = (KK + 31) & ~31;
    wq[i] = (unsigned short*)carve((size_t)wspec[i].O * kkpad[i] * 2);
    const int total = wspec[i].O * kkpad[i];
    prep_weight_kernel<<<(total + 255) / 256, 256, 0, stream>>>(
        wspec[i].w, wq[i], wspec[i].O, wspec[i].I, wspec[i].Kk, kkpad[i],
        wspec[i].isT);
  }
  prep_codebook_kernel<<<2, 256, 0, stream>>>(cb, cbq, cnorm);

  auto conv = [&](const float* srcf, const unsigned short* src16, int widx,
                  const float* bias, float* dstf, unsigned short* dst16,
                  int I, int Ti, int O, int To, int Kk, int sh, int padE,
                  int dl, int src_f32btf, int dst_fmt, int relu) {
    ConvArgs a;
    a.srcf = srcf; a.src16 = src16; a.bias = bias; a.wq = wq[widx];
    a.dstf = dstf; a.dst16 = dst16;
    a.I = I; a.Ti = Ti; a.O = O; a.To = To; a.Kk = Kk; a.KKpad = kkpad[widx];
    a.padE = padE; a.sh = sh; a.dl = dl;
    a.src_f32btf = src_f32btf; a.dst_fmt = dst_fmt; a.relu = relu;
    dim3 grid(To / 32, B, O / 256);
    conv1d_wmma_kernel<<<grid, 256, 0, stream>>>(a);
  };

  // ---- encoder (sh = log2 stride, dl = log2 lhs-dilation) ----
  conv(x, 0,       0, eb1, 0, bufA, F,   T,   256, 512, 3, 0, 1, 0, 1, 0, 1);
  conv(0, bufA,    1, eb2, 0, bufB, 256, 512, 256, 256, 4, 1, 1, 0, 0, 0, 1);
  conv(0, bufB,    2, eb3, 0, bufA, 256, 256, 512, 256, 3, 0, 1, 0, 0, 0, 1);
  conv(0, bufA,    3, eb4, 0, bufB, 512, 256, 512, 128, 4, 1, 1, 0, 0, 0, 1);
  conv(0, bufB,    4, eb5, zbuf, 0, 512, 128, 512, 128, 3, 0, 1, 0, 0, 1, 0);

  // ---- vector quantize ----
  dim3 vg(Tq / 16, B);
  vq_argmin_kernel<<<vg, 256, 0, stream>>>(zbuf, cbq, cnorm, idx_f, idxi, Tq);
  vq_gather_loss_kernel<<<B * Tq, 256, 0, stream>>>(zbuf, cb, idxi, zq16, part, Tq);
  vq_loss_finalize_kernel<<<1, 256, 0, stream>>>(part, loss, B * Tq);

  // ---- decoder (convT as lhs-dilated conv: sh=0, dl=1, padE=K-1-pad=2) ----
  conv(0, zq16,    5, db1,  0, bufA, 512, 128, 512, 128, 3, 0, 1, 0, 0, 0, 1);
  conv(0, bufA,    6, dbt2, 0, bufB, 512, 128, 512, 256, 4, 0, 2, 1, 0, 0, 1);
  conv(0, bufB,    7, db3,  0, bufA, 512, 256, 256, 256, 3, 0, 1, 0, 0, 0, 1);
  conv(0, bufA,    8, dbt4, 0, bufB, 256, 256, 256, 512, 4, 0, 2, 1, 0, 0, 1);
  conv(0, bufB,    9, db5,  recon, 0, 256, 512, 256, 512, 3, 0, 1, 0, 0, 2, 0);
}
